// DiffGNNLayer_31482110279902
// MI455X (gfx1250) — matrix operand order repaired and checked by reading, not compiled, not run on hardware
//
#include <hip/hip_runtime.h>
#include <hip/hip_bf16.h>
#include <math.h>

#define N_NODES 8192
#define N_EDGES 262144
#define ET      (N_EDGES + N_NODES)   // edges + self loops
#define DMODEL  32
#define X_LD    994

typedef __attribute__((ext_vector_type(16))) _Float16 v16h;
typedef __attribute__((ext_vector_type(8)))  _Float16 v8h;
typedef __attribute__((ext_vector_type(8)))  float    v8f;

// ---------------------------------------------------------------- utilities
__global__ void fill_f32(float* p, float v, int n) {
    int i = blockIdx.x * blockDim.x + threadIdx.x;
    if (i < n) p[i] = v;
}

__device__ inline void atomicMaxF(float* addr, float v) {
    if (v >= 0.f) atomicMax((int*)addr, __float_as_int(v));
    else          atomicMin((unsigned int*)addr, __float_as_uint(v));
}

// ---------------------------------------------------- Laplacian (en branch)
__global__ void en_extract(const float* __restrict__ x, float* __restrict__ en) {
    int i = blockIdx.x * blockDim.x + threadIdx.x;
    if (i < N_NODES) en[i] = x[i * X_LD + 97];
}

// dedup edges via N*N bitmap; build unique edge list (excluding diagonal) + out-degree
__global__ void dedup_edges(const int* __restrict__ src, const int* __restrict__ dst,
                            unsigned int* __restrict__ bitmap,
                            int* __restrict__ us, int* __restrict__ ud,
                            int* __restrict__ cnt, float* __restrict__ deg) {
    int e = blockIdx.x * blockDim.x + threadIdx.x;
    if (e >= N_EDGES) return;
    int s = src[e], d = dst[e];
    int bit = s * N_NODES + d;
    unsigned int mask = 1u << (bit & 31);
    unsigned int old = atomicOr(&bitmap[bit >> 5], mask);
    if (!(old & mask) && s != d) {
        int i = atomicAdd(cnt, 1);
        us[i] = s; ud[i] = d;
        atomicAdd(&deg[s], 1.0f);
    }
}

__global__ void lap_scatter(const int* __restrict__ us, const int* __restrict__ ud,
                            const int* __restrict__ cnt,
                            const float* __restrict__ h, float* __restrict__ acc) {
    int i = blockIdx.x * blockDim.x + threadIdx.x;
    if (i < *cnt) atomicAdd(&acc[us[i]], h[ud[i]]);
}

__global__ void lap_combine(const float* __restrict__ h, const float* __restrict__ acc,
                            const float* __restrict__ deg, float* __restrict__ out,
                            float scale, int n) {
    int i = blockIdx.x * blockDim.x + threadIdx.x;
    if (i < n) out[i] = (deg[i] * h[i] - acc[i]) * scale;
}

__global__ void en_proj(const float* __restrict__ h2, const float* __restrict__ W5,
                        const float* __restrict__ b5, float* __restrict__ hen) {
    int n = blockIdx.x, d = threadIdx.x;
    hen[n * DMODEL + d] = h2[n] * W5[d] + b5[d];
}

// ------------------------------------------------------------- small GEMMs
// Y[n][d] = (b?b[d]:0) + sum_k X[n*ldx + col0 + k] * W[k*dout + d]
__global__ void gemm_bias(const float* __restrict__ X, int ldx, int col0,
                          const float* __restrict__ W, const float* __restrict__ b,
                          float* __restrict__ Y, int K, int dout) {
    int n = blockIdx.x, d = threadIdx.x;
    const float* xr = X + (size_t)n * ldx + col0;
    float acc = b ? b[d] : 0.f;
    for (int k = 0; k < K; ++k) acc += xr[k] * W[k * dout + d];
    Y[n * dout + d] = acc;
}

// ---------------------------------------------------------------- GAT edges
__global__ void gat_attcoef(const float* __restrict__ h, const float* __restrict__ as_,
                            const float* __restrict__ ad_, float* __restrict__ hs,
                            float* __restrict__ hd, int dout) {
    int n = blockIdx.x * blockDim.x + threadIdx.x;
    if (n >= N_NODES) return;
    float s = 0.f, d = 0.f;
    for (int f = 0; f < dout; ++f) { float v = h[n * dout + f]; s += v * as_[f]; d += v * ad_[f]; }
    hs[n] = s; hd[n] = d;
}

__global__ void gat_edge1(const int* __restrict__ src0, const int* __restrict__ dst0,
                          const float* __restrict__ hs, const float* __restrict__ hd,
                          float* __restrict__ ew, float* __restrict__ m) {
    int e = blockIdx.x * blockDim.x + threadIdx.x;
    if (e >= ET) return;
    int s = (e < N_EDGES) ? src0[e] : (e - N_EDGES);
    int d = (e < N_EDGES) ? dst0[e] : (e - N_EDGES);
    float v = hs[s] + hd[d];
    v = v > 0.f ? v : 0.2f * v;           // leaky_relu 0.2
    ew[e] = v;
    atomicMaxF(&m[d], v);
}

__global__ void gat_edge2(const int* __restrict__ dst0, float* __restrict__ ew,
                          const float* __restrict__ m, float* __restrict__ wsum) {
    int e = blockIdx.x * blockDim.x + threadIdx.x;
    if (e >= ET) return;
    int d = (e < N_EDGES) ? dst0[e] : (e - N_EDGES);
    float w = __expf(ew[e] - m[d]);
    ew[e] = w;
    atomicAdd(&wsum[d], w);
}

__global__ void gat_edge3(const int* __restrict__ src0, const int* __restrict__ dst0,
                          const float* __restrict__ ew, const float* __restrict__ wsum,
                          const float* __restrict__ h, float* __restrict__ out, int dout) {
    int e = blockIdx.x;            // one block per edge
    int f = threadIdx.x;           // dout threads
    int s = (e < N_EDGES) ? src0[e] : (e - N_EDGES);
    int d = (e < N_EDGES) ? dst0[e] : (e - N_EDGES);
    float c = ew[e] / wsum[d];
    atomicAdd(&out[d * dout + f], c * h[s * dout + f]);
}

__global__ void add_bias(float* __restrict__ y, const float* __restrict__ b,
                         int dout, int total) {
    int i = blockIdx.x * blockDim.x + threadIdx.x;
    if (i < total) y[i] += b[i % dout];
}

// ------------------------------------------------------- cross-attn: QKV proj
// q,k stored row-major f16 (N x 32); v stored transposed f16 (32 x N) so all
// WMMA B-operand loads are contiguous.
__global__ void attn_proj(const float* __restrict__ xq, const float* __restrict__ y,
                          const float* __restrict__ Wq, const float* __restrict__ Wk,
                          const float* __restrict__ Wv,
                          _Float16* __restrict__ qh, _Float16* __restrict__ kh,
                          _Float16* __restrict__ vT) {
    int n = blockIdx.x, d = threadIdx.x;
    float q = 0.f, k = 0.f, v = 0.f;
    for (int t = 0; t < DMODEL; ++t) {
        float xv = xq[n * DMODEL + t], yv = y[n * DMODEL + t];
        q += xv * Wq[t * DMODEL + d];
        k += yv * Wk[t * DMODEL + d];
        v += yv * Wv[t * DMODEL + d];
    }
    qh[n * DMODEL + d] = (_Float16)q;
    kh[n * DMODEL + d] = (_Float16)k;
    vT[d * N_NODES + n] = (_Float16)v;
}

// --------------------------------------------- flash attention, WMMA 16x16x32
// One wave per 16 query rows. 128 keys per iteration: 8 score WMMAs, one
// online-softmax pass (shfl_xor butterflies amortized 4x vs 32-key blocks),
// P staged through LDS (D-layout -> A-layout), 8 PV WMMAs accumulate O.
// 1/sqrt(d) is folded into the A operand once.
__global__ __launch_bounds__(32)
void attn_fa(const _Float16* __restrict__ qh, const _Float16* __restrict__ kh,
             const _Float16* __restrict__ vT, float* __restrict__ out) {
    __shared__ _Float16 lds_p[16 * 128];
    const int lane = threadIdx.x;
    const int col  = lane & 15;
    const int hw   = lane >> 4;         // half-wave id
    const int q0   = blockIdx.x * 16;

    // A operand (16x32 f16): lane holds row=col; K = hw*16+{0..15}
    v8h qlo = *(const v8h*)(qh + (q0 + col) * DMODEL + hw * 16);
    v8h qhi = *(const v8h*)(qh + (q0 + col) * DMODEL + hw * 16 + 8);
    v16h aq = __builtin_shufflevector(qlo, qhi, 0,1,2,3,4,5,6,7,8,9,10,11,12,13,14,15);
    const _Float16 hscale = (_Float16)0.17677669529663687f;  // 1/sqrt(32)
#pragma unroll
    for (int i = 0; i < 16; ++i) aq[i] *= hscale;

    v8f o0 = {}; v8f o1 = {};
    float mrun[8], lrun[8];
#pragma unroll
    for (int r = 0; r < 8; ++r) { mrun[r] = -INFINITY; lrun[r] = 0.f; }

    for (int kb = 0; kb < N_NODES; kb += 128) {
        // ---- scores: 8 tiles of 16 keys
        v8f s[8];
#pragma unroll
        for (int t = 0; t < 8; ++t) {
            // B (32x16): lane holds key = kb+t*16+col, K(dims) = hw*16+{0..15}
            v16h bk = *(const v16h*)(kh + (kb + t * 16 + col) * DMODEL + hw * 16);
            v8f zc = {};
            s[t] = __builtin_amdgcn_wmma_f32_16x16x32_f16(false, aq, false, bk, (short)0, zc, false, false);
        }
        // ---- online softmax over 128 keys at once
#pragma unroll
        for (int r = 0; r < 8; ++r) {
            float mx = s[0][r];
#pragma unroll
            for (int t = 1; t < 8; ++t) mx = fmaxf(mx, s[t][r]);
#pragma unroll
            for (int msk = 1; msk <= 8; msk <<= 1) mx = fmaxf(mx, __shfl_xor(mx, msk, 32));
            float mn = fmaxf(mrun[r], mx);
            float alpha = __expf(mrun[r] - mn);
            float p[8];
            float rs = 0.f;
#pragma unroll
            for (int t = 0; t < 8; ++t) { p[t] = __expf(s[t][r] - mn); rs += p[t]; }
#pragma unroll
            for (int msk = 1; msk <= 8; msk <<= 1) rs += __shfl_xor(rs, msk, 32);
            lrun[r] = lrun[r] * alpha + rs;
            mrun[r] = mn;
            o0[r] *= alpha; o1[r] *= alpha;
            // stage P (16 rows x 128 keys, f16, row-major) into LDS
#pragma unroll
            for (int t = 0; t < 8; ++t)
                lds_p[(hw * 8 + r) * 128 + t * 16 + col] = (_Float16)p[t];
        }
        __syncthreads();
        // ---- PV: O(16x32) += P(16x128) @ V(128x32), 4 K-chunks of 32 keys
#pragma unroll
        for (int kc = 0; kc < 4; ++kc) {
            v8h plo = *(const v8h*)(lds_p + col * 128 + kc * 32 + hw * 16);
            v8h phi = *(const v8h*)(lds_p + col * 128 + kc * 32 + hw * 16 + 8);
            v16h ap = __builtin_shufflevector(plo, phi, 0,1,2,3,4,5,6,7,8,9,10,11,12,13,14,15);
            // B for PV (32 keys x 16 dims): contiguous because v is transposed
            v16h bv0 = *(const v16h*)(vT + (size_t)(col)      * N_NODES + kb + kc * 32 + hw * 16);
            v16h bv1 = *(const v16h*)(vT + (size_t)(16 + col) * N_NODES + kb + kc * 32 + hw * 16);
            o0 = __builtin_amdgcn_wmma_f32_16x16x32_f16(false, ap, false, bv0, (short)0, o0, false, false);
            o1 = __builtin_amdgcn_wmma_f32_16x16x32_f16(false, ap, false, bv1, (short)0, o1, false, false);
        }
        __syncthreads();
    }

#pragma unroll
    for (int r = 0; r < 8; ++r) {
        int row = q0 + hw * 8 + r;   // D layout: lanes 0-15 M=r, lanes 16-31 M=8+r
        out[row * DMODEL + col]      = o0[r] / lrun[r];
        out[row * DMODEL + col + 16] = o1[r] / lrun[r];
    }
}

// concat([attn_out, y]) @ Wl + bl
__global__ void attn_out_proj(const float* __restrict__ ao, const float* __restrict__ y,
                              const float* __restrict__ Wl, const float* __restrict__ bl,
                              float* __restrict__ out) {
    int n = blockIdx.x, d = threadIdx.x;
    float acc = bl[d];
    for (int k = 0; k < DMODEL; ++k) acc += ao[n * DMODEL + k] * Wl[k * DMODEL + d];
    for (int k = 0; k < DMODEL; ++k) acc += y[n * DMODEL + k] * Wl[(DMODEL + k) * DMODEL + d];
    out[n * DMODEL + d] = acc;
}

// ----------------------------------------------------------------- head MLPs
__global__ void fuse_w1(const float* __restrict__ he, const float* __restrict__ ha,
                        const float* __restrict__ hm, const float* __restrict__ hr,
                        const float* __restrict__ W1, const float* __restrict__ b1,
                        float* __restrict__ hsig) {
    int n = blockIdx.x, d = threadIdx.x;
    float acc = b1[d];
    for (int k = 0; k < 32; ++k) acc += he[n * 32 + k] * W1[k * 32 + d];
    for (int k = 0; k < 32; ++k) acc += ha[n * 32 + k] * W1[(32 + k) * 32 + d];
    for (int k = 0; k < 32; ++k) acc += hm[n * 32 + k] * W1[(64 + k) * 32 + d];
    for (int k = 0; k < 32; ++k) acc += hr[n * 32 + k] * W1[(96 + k) * 32 + d];
    hsig[n * 32 + d] = 1.f / (1.f + __expf(-acc));
}

__global__ void head(const float* __restrict__ hsig, const int* __restrict__ condp,
                     const float* __restrict__ W2, const float* __restrict__ b2,
                     const float* __restrict__ W3, const float* __restrict__ b3,
                     const int* __restrict__ y0, float* __restrict__ out) {
    int i = blockIdx.x * blockDim.x + threadIdx.x;
    if (i >= N_NODES - 1) return;
    int cond = *condp;
    int row = (i < cond) ? i : i + 1;
    float t16[16];
    for (int j = 0; j < 16; ++j) {
        float a = b2[j];
        for (int k = 0; k < 32; ++k) a += hsig[row  * 32 + k] * W2[k * 16 + j];
        for (int k = 0; k < 32; ++k) a += hsig[cond * 32 + k] * W2[(32 + k) * 16 + j];
        t16[j] = a;
    }
    float v = b3[0];
    for (int j = 0; j < 16; ++j) v += t16[j] * W3[j];
    out[i] = 1.f / (1.f + __expf(-v));
    float la = 0.f;
    for (int t = 0; t < 16; ++t) if (y0[t] == row) la = 1.f;
    out[(N_NODES - 1) + i] = la;
}

// ================================================================== launcher
extern "C" void kernel_launch(void* const* d_in, const int* in_sizes, int n_in,
                              void* d_out, int out_size, void* d_ws, size_t ws_size,
                              hipStream_t stream) {
    // input mapping (recursive insertion-order flatten of setup_inputs()):
    const float* x    = (const float*)d_in[0];
    const int*   edge = (const int*)d_in[1];
    const int*   src0 = edge;
    const int*   dst0 = edge + N_EDGES;
    const int*   y0   = (const int*)d_in[2];
    const int*   cond = (const int*)d_in[3];
    auto P = [&](int i) { return (const float*)d_in[i]; };
    // conv1..4 at 4,8,12,16 (W, a_s, a_d, b); ca1..3 at 20,25,30 (Wq,Wk,Wv,Wl,bl)
    const int IN_C1 = 4, IN_C2 = 8, IN_C3 = 12, IN_C4 = 16;
    const int IN_CA1 = 20, IN_CA2 = 25, IN_CA3 = 30;
    const int IN_W1 = 35, IN_B1 = 36, IN_W2 = 37, IN_B2 = 38, IN_W3 = 39, IN_B3 = 40;
    const int IN_W4 = 41, IN_B4 = 42, IN_W5 = 43, IN_B5 = 44;

    // ---- workspace partition (~24 MB)
    char* w = (char*)d_ws;
    auto alloc = [&](size_t bytes) -> char* {
        char* p = w; w += (bytes + 255) & ~(size_t)255; return p;
    };
    unsigned int* bitmap = (unsigned int*)alloc((size_t)N_NODES * N_NODES / 8);
    int*   us     = (int*)alloc(N_EDGES * 4);
    int*   ud     = (int*)alloc(N_EDGES * 4);
    int*   ucnt   = (int*)alloc(256);
    float* deg    = (float*)alloc(N_NODES * 4);
    float* lacc   = (float*)alloc(N_NODES * 4);
    float* enb    = (float*)alloc(N_NODES * 4);
    float* lh1    = (float*)alloc(N_NODES * 4);
    float* lh2    = (float*)alloc(N_NODES * 4);
    float* h_en   = (float*)alloc((size_t)N_NODES * 32 * 4);
    float* gat_h  = (float*)alloc((size_t)N_NODES * 32 * 4);
    float* gat_hs = (float*)alloc(N_NODES * 4);
    float* gat_hd = (float*)alloc(N_NODES * 4);
    float* gat_e  = (float*)alloc((size_t)ET * 4);
    float* gat_m  = (float*)alloc(N_NODES * 4);
    float* gat_ws = (float*)alloc(N_NODES * 4);
    float* gat_t  = (float*)alloc((size_t)N_NODES * 32 * 4);
    float* h_atom = (float*)alloc((size_t)N_NODES * 32 * 4);
    float* h_mot  = (float*)alloc((size_t)N_NODES * 32 * 4);
    float* h_rea  = (float*)alloc((size_t)N_NODES * 32 * 4);
    _Float16* qh  = (_Float16*)alloc((size_t)N_NODES * 32 * 2);
    _Float16* kh  = (_Float16*)alloc((size_t)N_NODES * 32 * 2);
    _Float16* vT  = (_Float16*)alloc((size_t)N_NODES * 32 * 2);
    float* attn_o = (float*)alloc((size_t)N_NODES * 32 * 4);
    float* h_ac   = (float*)alloc((size_t)N_NODES * 32 * 4);
    float* h_mc   = (float*)alloc((size_t)N_NODES * 32 * 4);
    float* h_rc   = (float*)alloc((size_t)N_NODES * 32 * 4);
    float* hsig   = (float*)alloc((size_t)N_NODES * 32 * 4);
    (void)ws_size; (void)n_in; (void)in_sizes; (void)out_size;

    const int TB = 256;
    const int GB_N  = (N_NODES + TB - 1) / TB;
    const int GB_E  = (N_EDGES + TB - 1) / TB;
    const int GB_ET = (ET + TB - 1) / TB;

    // ---- Laplacian branch: dedup adjacency, deg, (deg*h - A h) twice
    hipMemsetAsync(bitmap, 0, (size_t)N_NODES * N_NODES / 8, stream);
    hipMemsetAsync(ucnt, 0, 4, stream);
    hipMemsetAsync(deg, 0, N_NODES * 4, stream);
    en_extract<<<GB_N, TB, 0, stream>>>(x, enb);
    dedup_edges<<<GB_E, TB, 0, stream>>>(src0, dst0, bitmap, us, ud, ucnt, deg);
    hipMemsetAsync(lacc, 0, N_NODES * 4, stream);
    lap_scatter<<<GB_E, TB, 0, stream>>>(us, ud, ucnt, enb, lacc);
    lap_combine<<<GB_N, TB, 0, stream>>>(enb, lacc, deg, lh1, 1.0f, N_NODES);
    hipMemsetAsync(lacc, 0, N_NODES * 4, stream);
    lap_scatter<<<GB_E, TB, 0, stream>>>(us, ud, ucnt, lh1, lacc);
    lap_combine<<<GB_N, TB, 0, stream>>>(lh1, lacc, deg, lh2, 0.1f, N_NODES);
    en_proj<<<N_NODES, 32, 0, stream>>>(lh2, P(IN_W5), P(IN_B5), h_en);

    // ---- GAT layers
    auto run_gat = [&](const float* Xin, int ldx, int col0, int din, int dout,
                       int pb, float* outbuf) {
        gemm_bias<<<N_NODES, dout, 0, stream>>>(Xin, ldx, col0, P(pb), nullptr, gat_h, din, dout);
        gat_attcoef<<<GB_N, TB, 0, stream>>>(gat_h, P(pb + 1), P(pb + 2), gat_hs, gat_hd, dout);
        fill_f32<<<GB_N, TB, 0, stream>>>(gat_m, -INFINITY, N_NODES);
        hipMemsetAsync(gat_ws, 0, N_NODES * 4, stream);
        hipMemsetAsync(outbuf, 0, (size_t)N_NODES * dout * 4, stream);
        gat_edge1<<<GB_ET, TB, 0, stream>>>(src0, dst0, gat_hs, gat_hd, gat_e, gat_m);
        gat_edge2<<<GB_ET, TB, 0, stream>>>(dst0, gat_e, gat_m, gat_ws);
        gat_edge3<<<ET, dout, 0, stream>>>(src0, dst0, gat_e, gat_ws, gat_h, outbuf, dout);
        add_bias<<<(N_NODES * dout + TB - 1) / TB, TB, 0, stream>>>(outbuf, P(pb + 3), dout, N_NODES * dout);
    };
    run_gat(x, X_LD, 64, 33, 8,  IN_C1, gat_t);   // atom  33 -> 8
    run_gat(gat_t, 8, 0, 8, 32,  IN_C2, h_atom);  //        8 -> 32
    run_gat(x, X_LD, 0, 64, 8,   IN_C3, gat_t);   // motif 64 -> 8
    run_gat(gat_t, 8, 0, 8, 32,  IN_C4, h_mot);   //        8 -> 32

    // ---- reaction projection (896 -> 32)
    gemm_bias<<<N_NODES, 32, 0, stream>>>(x, X_LD, 98, P(IN_W4), P(IN_B4), h_rea, 896, 32);

    // ---- three cross-attentions (the WMMA hot path)
    auto run_ca = [&](const float* xq, const float* yv, int pb, float* outbuf) {
        attn_proj<<<N_NODES, 32, 0, stream>>>(xq, yv, P(pb), P(pb + 1), P(pb + 2), qh, kh, vT);
        attn_fa<<<N_NODES / 16, 32, 0, stream>>>(qh, kh, vT, attn_o);
        attn_out_proj<<<N_NODES, 32, 0, stream>>>(attn_o, yv, P(pb + 3), P(pb + 4), outbuf);
    };
    run_ca(h_en, h_atom, IN_CA1, h_ac);
    run_ca(h_ac, h_mot,  IN_CA2, h_mc);
    run_ca(h_mc, h_rea,  IN_CA3, h_rc);

    // ---- final MLPs + labels
    fuse_w1<<<N_NODES, 32, 0, stream>>>(h_en, h_ac, h_mc, h_rc, P(IN_W1), P(IN_B1), hsig);
    head<<<(N_NODES - 1 + TB - 1) / TB, TB, 0, stream>>>(hsig, cond, P(IN_W2), P(IN_B2),
                                                         P(IN_W3), P(IN_B3), y0, (float*)d_out);
}
